// SAGELayer_10101763080730
// MI455X (gfx1250) — matrix operand order repaired and checked by reading, not compiled, not run on hardware
//
#include <hip/hip_runtime.h>

// ---------------------------------------------------------------------------
// SAGE layer for MI455X (gfx1250), wave32 + v_wmma_f32_16x16x32_f16.
// Roofline: ~2.5GB HBM traffic (~110us @23.3TB/s) vs 210 GFLOP GEMM2
// (~40us at f16-WMMA rate) -> memory bound with f16 matrix math.
// GEMM2 B-matrix is staged in LDS (64KB/block) to cut L2 fragment traffic
// from 12.8GB to 1.6GB.
// ---------------------------------------------------------------------------

#define NN 100000
#define NE 1600000
#define DF 128      // node/edge feature dim, h dim
#define K1 256      // gemm1 K  (concat nf | mean-agg)
#define K2 256      // gemm2 K  (concat h[u] | h[v])
#define EO 256      // edge output dim
#define BSTRIDE 264 // LDS halves per B column: 256 + 8 pad (lane bank skew = 4)

typedef _Float16 half16 __attribute__((ext_vector_type(16)));
typedef _Float16 half8  __attribute__((ext_vector_type(8)));
typedef float    float8 __attribute__((ext_vector_type(8)));
typedef float    vfloat4 __attribute__((ext_vector_type(4)));

// ---------------- kernel 1: zero (msum region of d_out, cnt) ----------------
__global__ void zero_f32(float* __restrict__ p, int n) {
    int i = blockIdx.x * blockDim.x + threadIdx.x;
    int stride = gridDim.x * blockDim.x;
    for (; i < n; i += stride) p[i] = 0.0f;
}

// ---------------- kernel 2: convert + transpose weights to f16 --------------
// WaT[c*256 + k] = f16(W_apply_w[k*128 + c])   c<128, k<256  (col-major)
// WeT[c*256 + k] = f16(W_edge_w [k*256 + c])   c<256, k<256  (col-major)
__global__ void make_weights(const float* __restrict__ Wa,
                             const float* __restrict__ We,
                             _Float16* __restrict__ WaT,
                             _Float16* __restrict__ WeT) {
    int i = blockIdx.x * blockDim.x + threadIdx.x;
    if (i < DF * K1) {
        int c = i >> 8, k = i & 255;
        WaT[c * K1 + k] = (_Float16)Wa[k * DF + c];
    }
    if (i < EO * K2) {
        int c = i >> 8, k = i & 255;
        WeT[c * K2 + k] = (_Float16)We[k * EO + c];
    }
}

// ---------------- kernel 3: scatter-add edge feats + degree -----------------
__global__ void scatter_edges(const float* __restrict__ ef,
                              const int* __restrict__ v,
                              float* __restrict__ msum,
                              float* __restrict__ cnt) {
    unsigned int gid = blockIdx.x * blockDim.x + threadIdx.x; // < NE*128
    unsigned int e = gid >> 7;
    unsigned int d = gid & 127u;
    if (e >= NE) return;
    int dst = v[e];
    atomicAdd(&msum[(size_t)dst * DF + d], ef[(size_t)e * DF + d]);
    if (d == 0) atomicAdd(&cnt[dst], 1.0f);
}

// ---------------- kernel 4: GEMM1  h = relu([nf | msum/deg] @ Wa + ba) ------
// One wave per 16-row tile; N=128 (8 tiles of 16), K=256 (8 steps of 32).
// msum lives in the h region of d_out; reads finish before the tile's stores.
__global__ __launch_bounds__(256) void gemm1(const float* __restrict__ nf,
                                             const _Float16* __restrict__ WaT,
                                             const float* __restrict__ ba,
                                             const float* __restrict__ cnt,
                                             float* __restrict__ hmem,  // msum in, h out
                                             _Float16* __restrict__ H16) {
    int wave = (blockIdx.x * blockDim.x + threadIdx.x) >> 5;
    if (wave >= NN / 16) return;                 // wave-uniform guard
    int lane = threadIdx.x & 31;
    int row  = lane & 15;
    int hi   = lane >> 4;                        // 0 | 1
    const int koffA = hi ? 16 : 0;
    const int koffB = hi ? 16 : 0;
    const long rbase = (long)wave * 16;
    const long arow  = rbase + row;

    float rdeg = 1.0f / fmaxf(cnt[arow], 1.0f);

    float8 acc[8];
#pragma unroll
    for (int n = 0; n < 8; ++n) acc[n] = (float8)(0.0f);

#pragma unroll
    for (int ks = 0; ks < 8; ++ks) {
        const int kbase = ks * 32;
        const float* src;
        float scale;
        if (kbase < 128) { src = nf   + arow * DF + kbase + koffA;          scale = 1.0f; }
        else             { src = hmem + arow * DF + (kbase - 128) + koffA;  scale = rdeg; }
        vfloat4 f0 = *(const vfloat4*)(src);
        vfloat4 f1 = *(const vfloat4*)(src + 4);
        vfloat4 f2 = *(const vfloat4*)(src + 8);
        vfloat4 f3 = *(const vfloat4*)(src + 12);
        union { half16 v; _Float16 e[16]; } A;
#pragma unroll
        for (int i = 0; i < 4; ++i) {
            A.e[i]      = (_Float16)(f0[i] * scale);
            A.e[4 + i]  = (_Float16)(f1[i] * scale);
            A.e[8 + i]  = (_Float16)(f2[i] * scale);
            A.e[12 + i] = (_Float16)(f3[i] * scale);
        }
#pragma unroll
        for (int n = 0; n < 8; ++n) {
            const _Float16* bp = WaT + (size_t)(n * 16 + row) * K1 + kbase + koffB;
            half16 B = *(const half16*)bp;
            acc[n] = __builtin_amdgcn_wmma_f32_16x16x32_f16(
                false, A.v, false, B, (short)0, acc[n], false, false);
        }
    }
#pragma unroll
    for (int n = 0; n < 8; ++n) {
        int col = n * 16 + row;
        float bias = ba[col];
#pragma unroll
        for (int r = 0; r < 8; ++r) {
            long orow = rbase + r + (hi ? 8 : 0);
            float val = fmaxf(acc[n][r] + bias, 0.0f);
            hmem[orow * DF + col] = val;
            H16[orow * DF + col]  = (_Float16)val;
        }
    }
}

// ---------------- kernel 5: GEMM2  edge = [h[u] | h[v]] @ We + be -----------
// Block = 8 waves = 8 edge-tiles sharing one 128-col half of We, staged in
// LDS (64KB, padded for 4-bank lane skew). A gathered from f16 h (L2-hot).
// Grid is exactly (NE/16)*2/8 blocks: no guards, EXEC all-ones at WMMAs.
__global__ __launch_bounds__(256) void gemm2(const _Float16* __restrict__ H16,
                                             const int* __restrict__ u,
                                             const int* __restrict__ v,
                                             const _Float16* __restrict__ WeT,
                                             const float* __restrict__ be,
                                             float* __restrict__ eout) {
    __shared__ _Float16 Bsh[DF * BSTRIDE];       // 128 cols x (256+8) halves

    const int tid = threadIdx.x;
    const int nh  = blockIdx.x & 1;              // column half of We
    const long tileBase = (long)(blockIdx.x >> 1) * 8;

    // cooperative stage: 64 cols x 256 halves = 16384 halves, 64 per thread
#pragma unroll
    for (int j = 0; j < 8; ++j) {
        int chunk = tid + j * 256;               // 2048 chunks of 8 halves
        int c  = chunk >> 5;                     // 0..63
        int kc = (chunk & 31) * 8;               // 0..248
        *(half8*)&Bsh[c * BSTRIDE + kc] =
            *(const half8*)&WeT[(size_t)(nh * 128 + c) * K2 + kc];
    }
    __syncthreads();

    const int lane = tid & 31;
    const int row  = lane & 15;
    const int hi   = lane >> 4;
    const int koffA = hi ? 16 : 0;
    const int koffB = hi ? 16 : 0;
    const long tile = tileBase + (tid >> 5);

    long e  = tile * 16 + row;
    int un = u[e];
    int vn = v[e];

    float8 acc[8];
#pragma unroll
    for (int n = 0; n < 8; ++n) acc[n] = (float8)(0.0f);

#pragma unroll
    for (int ks = 0; ks < 8; ++ks) {
        int node = (ks < 4) ? un : vn;
        int kk   = (ks & 3) * 32 + koffA;
        const _Float16* ap = H16 + (size_t)node * DF + kk;
        union { half16 v; half8 h[2]; } A;
        A.h[0] = *(const half8*)(ap);
        A.h[1] = *(const half8*)(ap + 8);
#pragma unroll
        for (int n = 0; n < 8; ++n) {
            const _Float16* bp = &Bsh[(n * 16 + row) * BSTRIDE + ks * 32 + koffB];
            union { half16 v; half8 h[2]; } B;
            B.h[0] = *(const half8*)(bp);
            B.h[1] = *(const half8*)(bp + 8);
            acc[n] = __builtin_amdgcn_wmma_f32_16x16x32_f16(
                false, A.v, false, B.v, (short)0, acc[n], false, false);
        }
    }
#pragma unroll
    for (int n = 0; n < 8; ++n) {
        int col = nh * 128 + n * 16 + row;
        float bias = be[col];
#pragma unroll
        for (int r = 0; r < 8; ++r) {
            long orow = tile * 16 + r + (hi ? 8 : 0);
            eout[orow * EO + col] = acc[n][r] + bias;
        }
    }
}

// ---------------------------------------------------------------------------
extern "C" void kernel_launch(void* const* d_in, const int* in_sizes, int n_in,
                              void* d_out, int out_size, void* d_ws, size_t ws_size,
                              hipStream_t stream) {
    const float* nf = (const float*)d_in[0];
    const float* ef = (const float*)d_in[1];
    const int*   u  = (const int*)d_in[2];
    const int*   v  = (const int*)d_in[3];
    const float* Wa = (const float*)d_in[4];
    const float* ba = (const float*)d_in[5];
    const float* We = (const float*)d_in[6];
    const float* be = (const float*)d_in[7];

    float* hout = (float*)d_out;                       // N*128 f32 (msum scratch, then h)
    float* eout = hout + (size_t)NN * DF;              // E*256 f32

    // workspace layout
    char* ws = (char*)d_ws;
    size_t off = 0;
    float* cnt = (float*)(ws + off);  off += (size_t)NN * sizeof(float);
    off = (off + 255) & ~(size_t)255;
    _Float16* H16 = (_Float16*)(ws + off); off += (size_t)NN * DF * sizeof(_Float16);
    _Float16* WaT = (_Float16*)(ws + off); off += (size_t)DF * K1 * sizeof(_Float16);
    _Float16* WeT = (_Float16*)(ws + off); off += (size_t)EO * K2 * sizeof(_Float16);
    (void)ws_size; (void)in_sizes; (void)n_in; (void)out_size;

    // 1. zero msum (h region of d_out) and degree counters
    zero_f32<<<4096, 256, 0, stream>>>(hout, NN * DF);
    zero_f32<<<256, 256, 0, stream>>>(cnt, NN);

    // 2. f16 transposed weights (tiny; L2 resident afterwards)
    make_weights<<<(EO * K2 + 255) / 256, 256, 0, stream>>>(Wa, We, WaT, WeT);

    // 3. scatter-add 1.6M x 128 edge features (f32 atomics into 51MB -> L2)
    {
        unsigned int total = (unsigned int)NE * 128u;
        scatter_edges<<<total / 256, 256, 0, stream>>>(ef, v, hout, cnt);
    }

    // 4. GEMM1: 6250 row-tiles, one wave each (8 waves / 256-thread block)
    gemm1<<<(NN / 16 + 7) / 8, 256, 0, stream>>>(nf, WaT, ba, cnt, hout, H16);

    // 5. GEMM2: 100000 edge-tiles x 2 column-halves; 8 waves/block share LDS B
    gemm2<<<(NE / 16) * 2 / 8, 256, 0, stream>>>(H16, u, v, WeT, be, eout);
}
